// split_GPSA_55791625175625
// MI455X (gfx1250) — matrix operand order repaired and checked by reading, not compiled, not run on hardware
//
#include <hip/hip_runtime.h>
#include <math.h>

// ---------------------------------------------------------------------------
// ConViT GPSA forward for MI455X (gfx1250, wave32, WMMA).
// All GEMM-like work runs on v_wmma_f32_16x16x32_bf16 (f32 accumulate).
// ---------------------------------------------------------------------------

typedef __attribute__((ext_vector_type(16))) __bf16 v16bf;
typedef __attribute__((ext_vector_type(8)))  float  v8f;

#define WMMA_BF16(A, B, Cacc) \
  __builtin_amdgcn_wmma_f32_16x16x32_bf16(false, (A), false, (B), (short)0, (Cacc), false, false)

union BPack {
  v16bf v;
  unsigned int u[8];
  unsigned short h[16];
};

__device__ __forceinline__ unsigned short f2bf(float f) {
  union { float f; unsigned int u; } cv; cv.f = f;
  unsigned int r = cv.u + 0x7FFFu + ((cv.u >> 16) & 1u);  // round-to-nearest-even
  return (unsigned short)(r >> 16);
}

__device__ __forceinline__ v8f vzero8() {
  v8f z = {0.f, 0.f, 0.f, 0.f, 0.f, 0.f, 0.f, 0.f};
  return z;
}

// A-tile loader for 16x64 (d=48 zero-padded) operand from a row-major [n][48]
// bf16 buffer.  A layout (16-bit 16x32): lanes 0-15: M=l, K=0..7 & 16..23;
// lanes 16-31: M=l-16, K=8..15 & 24..31.  Second tile (K=32..63): elements
// 8..15 map to d>=48 -> zero for every lane (uniform).
__device__ __forceinline__ void load_a_k48(const unsigned short* __restrict__ row,
                                           int half, BPack& a0, BPack& a1) {
  uint4 t0 = *(const uint4*)(row + 8 * half);
  uint4 t1 = *(const uint4*)(row + 16 + 8 * half);
  a0.u[0] = t0.x; a0.u[1] = t0.y; a0.u[2] = t0.z; a0.u[3] = t0.w;
  a0.u[4] = t1.x; a0.u[5] = t1.y; a0.u[6] = t1.z; a0.u[7] = t1.w;
  uint4 t2 = *(const uint4*)(row + 32 + 8 * half);
  a1.u[0] = t2.x; a1.u[1] = t2.y; a1.u[2] = t2.z; a1.u[3] = t2.w;
  a1.u[4] = 0u;   a1.u[5] = 0u;   a1.u[6] = 0u;   a1.u[7] = 0u;
}

// B-tile loader for 64x16 (d=48 zero-padded) operand, contraction over d, from
// a row-major [n][48] bf16 buffer.  B layout matches A: lanes 0-15 hold
// K=0..7 & 16..23, lanes 16-31 hold K=8..15 & 24..31.  Second tile
// (K=32..63): elements 8..15 map to d>=48 -> zero.
__device__ __forceinline__ void load_b_q48(const unsigned short* __restrict__ row,
                                           int half, BPack& b0, BPack& b1) {
  uint4 t0 = *(const uint4*)(row + 8 * half);
  uint4 t1 = *(const uint4*)(row + 16 + 8 * half);
  b0.u[0] = t0.x; b0.u[1] = t0.y; b0.u[2] = t0.z; b0.u[3] = t0.w;
  b0.u[4] = t1.x; b0.u[5] = t1.y; b0.u[6] = t1.z; b0.u[7] = t1.w;
  uint4 t2 = *(const uint4*)(row + 32 + 8 * half);
  b1.u[0] = t2.x; b1.u[1] = t2.y; b1.u[2] = t2.z; b1.u[3] = t2.w;
  b1.u[4] = 0u;   b1.u[5] = 0u;   b1.u[6] = 0u;   b1.u[7] = 0u;
}

// P(16x32, A layout) x V-tile(32x16) -> O accumulate.  V is stored transposed
// ([bh][48][784]) so B dword pairs (K=2r,2r+1) are contiguous in memory.
__device__ __forceinline__ v8f pv_wmma(v16bf pa, const unsigned short* __restrict__ vrow,
                                       v8f O) {
  uint4 v0 = *(const uint4*)(vrow);
  uint4 v1 = *(const uint4*)(vrow + 16);
  BPack bv;
  bv.u[0] = v0.x; bv.u[1] = v0.y; bv.u[2] = v0.z; bv.u[3] = v0.w;
  bv.u[4] = v1.x; bv.u[5] = v1.y; bv.u[6] = v1.z; bv.u[7] = v1.w;
  return WMMA_BF16(pa, bv.v, O);
}

// ---------------------------------------------------------------------------
// Kernel 1: f32 -> bf16 conversion (grid-stride).
// ---------------------------------------------------------------------------
__global__ void cvt_bf16_kernel(const float* __restrict__ src,
                                unsigned short* __restrict__ dst, int n) {
  int i = blockIdx.x * blockDim.x + threadIdx.x;
  int st = gridDim.x * blockDim.x;
  for (; i < n; i += st) dst[i] = f2bf(src[i]);
}

// ---------------------------------------------------------------------------
// Kernel 2: positional softmax, batch-independent.  One block per (h, n) row.
// Writes posg[h][n][m] = sigmoid(gating[h]) * softmax_m(pos_score).
// ---------------------------------------------------------------------------
__global__ void __launch_bounds__(256) pos_softmax_kernel(
    const float* __restrict__ Wpos, const float* __restrict__ bpos,
    const float* __restrict__ gating, float* __restrict__ posg) {
  const int row = blockIdx.x;              // h*784 + n
  const int h = row / 784, n = row % 784;
  const int nx = n % 28, ny = n / 28;
  const float w0 = Wpos[h * 3 + 0], w1 = Wpos[h * 3 + 1], w2 = Wpos[h * 3 + 2];
  const float bb = bpos[h];
  const float g = 1.f / (1.f + __expf(-gating[h]));

  __shared__ float red[256];
  float v[4];
  int cols[4];
  int cnt = 0;
  float lmax = -1e30f;
  for (int m = threadIdx.x; m < 784; m += 256) {
    int mx = m % 28, my = m / 28;
    float dx = (float)(mx - nx), dy = (float)(my - ny);
    float sc = dx * w0 + dy * w1 + (dx * dx + dy * dy) * w2 + bb;
    v[cnt] = sc; cols[cnt] = m; ++cnt;
    lmax = fmaxf(lmax, sc);
  }
  red[threadIdx.x] = lmax;
  __syncthreads();
  for (int s = 128; s > 0; s >>= 1) {
    if ((int)threadIdx.x < s) red[threadIdx.x] = fmaxf(red[threadIdx.x], red[threadIdx.x + s]);
    __syncthreads();
  }
  const float rmax = red[0];
  __syncthreads();
  float lsum = 0.f;
  for (int i = 0; i < cnt; ++i) { v[i] = __expf(v[i] - rmax); lsum += v[i]; }
  red[threadIdx.x] = lsum;
  __syncthreads();
  for (int s = 128; s > 0; s >>= 1) {
    if ((int)threadIdx.x < s) red[threadIdx.x] += red[threadIdx.x + s];
    __syncthreads();
  }
  const float inv = g / red[0];
  for (int i = 0; i < cnt; ++i)
    posg[(size_t)row * 784 + cols[i]] = v[i] * inv;
}

// ---------------------------------------------------------------------------
// Kernel 3: QKV projections, y = x @ W^T.  Each wave computes a 16x64 tile of
// one of the three outputs.  A = x (row-major bf16), B = W^T (pairs contiguous
// along c in row-major W).  Writes q/k row-major [bh][n][48] (q pre-scaled by
// d^-1/2), v transposed [bh][48][784] (bf16) plus the f32 v output.
// ---------------------------------------------------------------------------
__global__ void __launch_bounds__(256) qkv_wmma_kernel(
    const unsigned short* __restrict__ xb,
    const unsigned short* __restrict__ wq,
    const unsigned short* __restrict__ wk,
    const unsigned short* __restrict__ wv,
    unsigned short* __restrict__ qbuf,
    unsigned short* __restrict__ kbuf,
    unsigned short* __restrict__ vbt,
    float* __restrict__ vout) {
  const int lane = threadIdx.x & 31;
  const int l16 = lane & 15;
  const int half = lane >> 4;
  const int task = blockIdx.x * 8 + (threadIdx.x >> 5);
  if (task >= 3 * 392 * 12) return;
  const int w = task / (392 * 12);
  const int rem = task % (392 * 12);
  const int row0 = (rem / 12) * 16;
  const int col0 = (rem % 12) * 64;
  const unsigned short* W = (w == 0) ? wq : (w == 1) ? wk : wv;

  v8f acc0 = vzero8(), acc1 = vzero8(), acc2 = vzero8(), acc3 = vzero8();
  const unsigned short* arow = xb + (size_t)(row0 + l16) * 768;
  const unsigned short* br0 = W + (size_t)(col0 + 0 + l16) * 768;
  const unsigned short* br1 = W + (size_t)(col0 + 16 + l16) * 768;
  const unsigned short* br2 = W + (size_t)(col0 + 32 + l16) * 768;
  const unsigned short* br3 = W + (size_t)(col0 + 48 + l16) * 768;

  for (int kk = 0; kk < 768; kk += 32) {
    BPack a;
    uint4 t0 = *(const uint4*)(arow + kk + 8 * half);
    uint4 t1 = *(const uint4*)(arow + kk + 16 + 8 * half);
    a.u[0] = t0.x; a.u[1] = t0.y; a.u[2] = t0.z; a.u[3] = t0.w;
    a.u[4] = t1.x; a.u[5] = t1.y; a.u[6] = t1.z; a.u[7] = t1.w;
    __builtin_prefetch(arow + kk + 64, 0, 0);  // global_prefetch_b8

    BPack b;
#define LOADB_MM(rowptr, accv)                                        \
    {                                                                 \
      uint4 c0 = *(const uint4*)((rowptr) + kk + 8 * half);           \
      uint4 c1 = *(const uint4*)((rowptr) + kk + 16 + 8 * half);      \
      b.u[0] = c0.x; b.u[1] = c0.y; b.u[2] = c0.z; b.u[3] = c0.w;     \
      b.u[4] = c1.x; b.u[5] = c1.y; b.u[6] = c1.z; b.u[7] = c1.w;     \
      accv = WMMA_BF16(a.v, b.v, accv);                               \
    }
    LOADB_MM(br0, acc0)
    LOADB_MM(br1, acc1)
    LOADB_MM(br2, acc2)
    LOADB_MM(br3, acc3)
#undef LOADB_MM
  }

  const float qscale = 0.14433756729740643f;  // 48^-0.5 folded into Q
  v8f accs[4] = {acc0, acc1, acc2, acc3};
#pragma unroll
  for (int c = 0; c < 4; ++c) {
#pragma unroll
    for (int r = 0; r < 8; ++r) {
      const int m = r + 8 * half;          // D layout: M = r + 8*(lane/16)
      const int row = row0 + m;
      const int bb = row / 784, n = row % 784;
      const int j = col0 + c * 16 + l16;   // D layout: N = lane%16
      const int hh = j / 48, d = j % 48;
      const size_t bh = (size_t)bb * 16 + hh;
      const float val = accs[c][r];
      if (w == 0) {
        qbuf[(bh * 784 + n) * 48 + d] = f2bf(val * qscale);
      } else if (w == 1) {
        kbuf[(bh * 784 + n) * 48 + d] = f2bf(val);
      } else {
        vbt[(bh * 48 + d) * 784 + n] = f2bf(val);
        vout[(bh * 784 + n) * 48 + d] = val;
      }
    }
  }
}

// ---------------------------------------------------------------------------
// Kernel 4: content-score row stats (online max / sum-exp).  Each wave owns 16
// queries of one (b,h): computes S^T = K.Q^T tile-by-tile; each lane tracks
// stats for its own query over the keys it sees, halves merged via shfl_xor 16.
// ---------------------------------------------------------------------------
__global__ void __launch_bounds__(256) attn_stats_kernel(
    const unsigned short* __restrict__ qbuf,
    const unsigned short* __restrict__ kbuf,
    float* __restrict__ rowmax, float* __restrict__ rowsum) {
  const int lane = threadIdx.x & 31;
  const int l16 = lane & 15;
  const int half = lane >> 4;
  const int task = blockIdx.x * 8 + (threadIdx.x >> 5);
  if (task >= 128 * 49) return;
  const int bh = task / 49;
  const int q0 = (task % 49) * 16;
  const int nq = q0 + l16;

  BPack b0, b1;
  load_b_q48(qbuf + ((size_t)bh * 784 + nq) * 48, half, b0, b1);

  float rm = -1e30f, rs = 0.f;
  for (int kt = 0; kt < 49; ++kt) {
    BPack a0, a1;
    load_a_k48(kbuf + ((size_t)bh * 784 + kt * 16 + l16) * 48, half, a0, a1);
    v8f c = vzero8();
    c = WMMA_BF16(a0.v, b0.v, c);
    c = WMMA_BF16(a1.v, b1.v, c);
    // lane holds keys (kt*16 + r + 8*half) for query nq
    float tm = c[0];
#pragma unroll
    for (int r = 1; r < 8; ++r) tm = fmaxf(tm, c[r]);
    const float nm = fmaxf(rm, tm);
    const float resc = __expf(rm - nm);
    float acc = 0.f;
#pragma unroll
    for (int r = 0; r < 8; ++r) acc += __expf(c[r] - nm);
    rs = rs * resc + acc;
    rm = nm;
  }
  const float om = __shfl_xor(rm, 16, 32);
  const float os = __shfl_xor(rs, 16, 32);
  const float nm = fmaxf(rm, om);
  const float fs = rs * __expf(rm - nm) + os * __expf(om - nm);
  if (half == 0) {
    rowmax[(size_t)bh * 784 + nq] = nm;
    rowsum[(size_t)bh * 784 + nq] = fs;
  }
}

// ---------------------------------------------------------------------------
// Kernel 5: attention apply.  Recomputes S^T tiles, forms
// attn = (1-g)*softmax(content) + g*softmax(pos)  (combined row-sum == 1),
// streams attn to global (float4), and feeds the combined probabilities
// straight into P@V WMMAs: the S^T D-layout per lane *is* the A layout for
// the next WMMA (pair two 16-key tiles for K=32).  Head outputs -> bf16 hob.
// ---------------------------------------------------------------------------
__global__ void __launch_bounds__(256) attn_apply_kernel(
    const unsigned short* __restrict__ qbuf,
    const unsigned short* __restrict__ kbuf,
    const unsigned short* __restrict__ vbt,
    const float* __restrict__ rowmax,
    const float* __restrict__ rowsum,
    const float* __restrict__ posg,
    const float* __restrict__ gating,
    float* __restrict__ attn_out,
    unsigned short* __restrict__ hob) {
  const int lane = threadIdx.x & 31;
  const int l16 = lane & 15;
  const int half = lane >> 4;
  const int task = blockIdx.x * 8 + (threadIdx.x >> 5);
  if (task >= 128 * 49) return;
  const int bh = task / 49;
  const int q0 = (task % 49) * 16;
  const int nq = q0 + l16;
  const int b = bh >> 4;
  const int h = bh & 15;

  BPack b0, b1;
  load_b_q48(qbuf + ((size_t)bh * 784 + nq) * 48, half, b0, b1);

  const float rm = rowmax[(size_t)bh * 784 + nq];
  const float inv = 1.f / rowsum[(size_t)bh * 784 + nq];
  const float g = 1.f / (1.f + __expf(-gating[h]));
  const float cg = 1.f - g;

  const float* prow = posg + ((size_t)h * 784 + nq) * 784;
  float* arow_out = attn_out + ((size_t)bh * 784 + nq) * 784;
  const unsigned short* vbase = vbt + (size_t)bh * 48 * 784;

  v8f O0 = vzero8(), O1 = vzero8(), O2 = vzero8();
  float prev[8];

  for (int kt = 0; kt < 49; ++kt) {
    BPack a0, a1;
    load_a_k48(kbuf + ((size_t)bh * 784 + kt * 16 + l16) * 48, half, a0, a1);
    v8f c = vzero8();
    c = WMMA_BF16(a0.v, b0.v, c);
    c = WMMA_BF16(a1.v, b1.v, c);

    const int kb16 = kt * 16 + 8 * half;   // lane's 8 consecutive keys
    const float4 p0 = *(const float4*)(prow + kb16);
    const float4 p1 = *(const float4*)(prow + kb16 + 4);
    float comb[8];
    comb[0] = cg * __expf(c[0] - rm) * inv + p0.x;
    comb[1] = cg * __expf(c[1] - rm) * inv + p0.y;
    comb[2] = cg * __expf(c[2] - rm) * inv + p0.z;
    comb[3] = cg * __expf(c[3] - rm) * inv + p0.w;
    comb[4] = cg * __expf(c[4] - rm) * inv + p1.x;
    comb[5] = cg * __expf(c[5] - rm) * inv + p1.y;
    comb[6] = cg * __expf(c[6] - rm) * inv + p1.z;
    comb[7] = cg * __expf(c[7] - rm) * inv + p1.w;

    float4 s0 = make_float4(comb[0], comb[1], comb[2], comb[3]);
    float4 s1 = make_float4(comb[4], comb[5], comb[6], comb[7]);
    *(float4*)(arow_out + kb16) = s0;
    *(float4*)(arow_out + kb16 + 4) = s1;

    if ((kt & 1) == 0) {
#pragma unroll
      for (int r = 0; r < 8; ++r) prev[r] = comb[r];
    } else {
      BPack pa;  // A tile: K = keys [16*(kt-1), 16*(kt-1)+31]
#pragma unroll
      for (int e = 0; e < 8; ++e) pa.h[e] = f2bf(prev[e]);
#pragma unroll
      for (int e = 0; e < 8; ++e) pa.h[8 + e] = f2bf(comb[e]);
      const int kpb = (kt - 1) * 16 + 8 * half;
      O0 = pv_wmma(pa.v, vbase + (size_t)(0 + l16) * 784 + kpb, O0);
      O1 = pv_wmma(pa.v, vbase + (size_t)(16 + l16) * 784 + kpb, O1);
      O2 = pv_wmma(pa.v, vbase + (size_t)(32 + l16) * 784 + kpb, O2);
    }
  }
  {  // leftover tile kt=48 paired with zeros (vbt is padded past row end)
    BPack pa;
#pragma unroll
    for (int e = 0; e < 8; ++e) pa.h[e] = f2bf(prev[e]);
    pa.u[4] = 0u; pa.u[5] = 0u; pa.u[6] = 0u; pa.u[7] = 0u;
    const int kpb = 768 + 8 * half;
    O0 = pv_wmma(pa.v, vbase + (size_t)(0 + l16) * 784 + kpb, O0);
    O1 = pv_wmma(pa.v, vbase + (size_t)(16 + l16) * 784 + kpb, O1);
    O2 = pv_wmma(pa.v, vbase + (size_t)(32 + l16) * 784 + kpb, O2);
  }

  // O D-layout: M = query-in-tile = r + 8*half, N = d = dt*16 + lane%16.
#pragma unroll
  for (int r = 0; r < 8; ++r) {
    const size_t row = (size_t)b * 784 + q0 + r + 8 * half;
    hob[row * 768 + h * 48 + 0 + l16] = f2bf(O0[r]);
    hob[row * 768 + h * 48 + 16 + l16] = f2bf(O1[r]);
    hob[row * 768 + h * 48 + 32 + l16] = f2bf(O2[r]);
  }
}

// ---------------------------------------------------------------------------
// Kernel 6: output projection out = HO @ Wproj^T + b_proj (f32 out).
// ---------------------------------------------------------------------------
__global__ void __launch_bounds__(256) proj_wmma_kernel(
    const unsigned short* __restrict__ hob,
    const unsigned short* __restrict__ wp,
    const float* __restrict__ bias,
    float* __restrict__ out) {
  const int lane = threadIdx.x & 31;
  const int l16 = lane & 15;
  const int half = lane >> 4;
  const int task = blockIdx.x * 8 + (threadIdx.x >> 5);
  if (task >= 392 * 12) return;
  const int row0 = (task / 12) * 16;
  const int col0 = (task % 12) * 64;

  v8f acc0 = vzero8(), acc1 = vzero8(), acc2 = vzero8(), acc3 = vzero8();
  const unsigned short* arow = hob + (size_t)(row0 + l16) * 768;
  const unsigned short* br0 = wp + (size_t)(col0 + 0 + l16) * 768;
  const unsigned short* br1 = wp + (size_t)(col0 + 16 + l16) * 768;
  const unsigned short* br2 = wp + (size_t)(col0 + 32 + l16) * 768;
  const unsigned short* br3 = wp + (size_t)(col0 + 48 + l16) * 768;

  for (int kk = 0; kk < 768; kk += 32) {
    BPack a;
    uint4 t0 = *(const uint4*)(arow + kk + 8 * half);
    uint4 t1 = *(const uint4*)(arow + kk + 16 + 8 * half);
    a.u[0] = t0.x; a.u[1] = t0.y; a.u[2] = t0.z; a.u[3] = t0.w;
    a.u[4] = t1.x; a.u[5] = t1.y; a.u[6] = t1.z; a.u[7] = t1.w;
    __builtin_prefetch(arow + kk + 64, 0, 0);

    BPack b;
#define LOADB_MM(rowptr, accv)                                        \
    {                                                                 \
      uint4 c0 = *(const uint4*)((rowptr) + kk + 8 * half);           \
      uint4 c1 = *(const uint4*)((rowptr) + kk + 16 + 8 * half);      \
      b.u[0] = c0.x; b.u[1] = c0.y; b.u[2] = c0.z; b.u[3] = c0.w;     \
      b.u[4] = c1.x; b.u[5] = c1.y; b.u[6] = c1.z; b.u[7] = c1.w;     \
      accv = WMMA_BF16(a.v, b.v, accv);                               \
    }
    LOADB_MM(br0, acc0)
    LOADB_MM(br1, acc1)
    LOADB_MM(br2, acc2)
    LOADB_MM(br3, acc3)
#undef LOADB_MM
  }

  v8f accs[4] = {acc0, acc1, acc2, acc3};
#pragma unroll
  for (int c = 0; c < 4; ++c) {
#pragma unroll
    for (int r = 0; r < 8; ++r) {
      const int m = r + 8 * half;
      const int j = col0 + c * 16 + l16;
      out[(size_t)(row0 + m) * 768 + j] = accs[c][r] + bias[j];
    }
  }
}

// ---------------------------------------------------------------------------
// Host launch.
// ---------------------------------------------------------------------------
extern "C" void kernel_launch(void* const* d_in, const int* in_sizes, int n_in,
                              void* d_out, int out_size, void* d_ws, size_t ws_size,
                              hipStream_t stream) {
  (void)in_sizes; (void)n_in; (void)out_size; (void)ws_size;

  const float* x      = (const float*)d_in[0];
  const float* Wq     = (const float*)d_in[1];
  const float* Wk     = (const float*)d_in[2];
  const float* Wv     = (const float*)d_in[3];
  const float* Wproj  = (const float*)d_in[4];
  const float* bproj  = (const float*)d_in[5];
  const float* Wpos   = (const float*)d_in[6];
  const float* bpos   = (const float*)d_in[7];
  const float* gating = (const float*)d_in[8];

  // Output regions (return order: out, attn, v)
  float* out_ptr  = (float*)d_out;
  float* attn_ptr = out_ptr + (size_t)8 * 784 * 768;
  float* v_ptr    = attn_ptr + (size_t)8 * 16 * 784 * 784;

  // Workspace layout (~93 MB)
  unsigned short* p = (unsigned short*)d_ws;
  unsigned short* xb   = p; p += (size_t)6272 * 768;       // x bf16
  unsigned short* wqb  = p; p += (size_t)768 * 768;
  unsigned short* wkb  = p; p += (size_t)768 * 768;
  unsigned short* wvb  = p; p += (size_t)768 * 768;
  unsigned short* wpb  = p; p += (size_t)768 * 768;
  unsigned short* qbuf = p; p += (size_t)128 * 784 * 48;   // Q*scale bf16 [bh][n][d]
  unsigned short* kbuf = p; p += (size_t)128 * 784 * 48;   // K bf16 [bh][n][d]
  unsigned short* vbt  = p; p += (size_t)128 * 48 * 784 + 1024;  // V^T bf16 (+pad)
  unsigned short* hob  = p; p += (size_t)6272 * 768;       // head outputs bf16
  float* posg   = (float*)p;                               // g*softmax(pos) [16][784][784]
  float* rowmax = posg + (size_t)16 * 784 * 784;
  float* rowsum = rowmax + (size_t)128 * 784;

  cvt_bf16_kernel<<<1024, 256, 0, stream>>>(x, xb, 6272 * 768);
  cvt_bf16_kernel<<<288, 256, 0, stream>>>(Wq, wqb, 768 * 768);
  cvt_bf16_kernel<<<288, 256, 0, stream>>>(Wk, wkb, 768 * 768);
  cvt_bf16_kernel<<<288, 256, 0, stream>>>(Wv, wvb, 768 * 768);
  cvt_bf16_kernel<<<288, 256, 0, stream>>>(Wproj, wpb, 768 * 768);

  pos_softmax_kernel<<<16 * 784, 256, 0, stream>>>(Wpos, bpos, gating, posg);

  // 3 weights * 392 row-tiles * 12 col-groups = 14112 wave-tasks / 8 waves
  qkv_wmma_kernel<<<1764, 256, 0, stream>>>(xb, wqb, wkb, wvb, qbuf, kbuf, vbt, v_ptr);

  // 128 (b,h) * 49 query-tiles = 6272 wave-tasks / 8 waves
  attn_stats_kernel<<<784, 256, 0, stream>>>(qbuf, kbuf, rowmax, rowsum);
  attn_apply_kernel<<<784, 256, 0, stream>>>(qbuf, kbuf, vbt, rowmax, rowsum,
                                             posg, gating, attn_ptr, hob);

  // 392 * 12 = 4704 wave-tasks / 8 waves
  proj_wmma_kernel<<<588, 256, 0, stream>>>(hob, wpb, bproj, out_ptr);
}
